// Hmoe_43379169690208
// MI455X (gfx1250) — matrix operand order repaired and checked by reading, not compiled, not run on hardware
//
#include <hip/hip_runtime.h>

// ---------------------------------------------------------------------------
// HMoE (soft decision tree) fused kernel for MI455X (gfx1250, wave32).
//   gate = sigmoid(x @ w + b); running = tree path products; out = running@resp
// Both GEMMs via v_wmma_f32_16x16x32_bf16 (f32 accumulate), LDS double-buffered.
// ---------------------------------------------------------------------------

typedef __attribute__((ext_vector_type(16))) __bf16 v16bf;
typedef __attribute__((ext_vector_type(8)))  float  v8f;

union Frag16 { v16bf v; uint4 q[2]; unsigned short u[16]; };

#define DIMX 2048
#define DIMY 2048
#define LVS  64     // leaves
#define TM   128    // rows per workgroup

// Pack two f32 -> packed bf16 pair (round-half-up) with a single v_perm_b32.
__device__ __forceinline__ unsigned int pack2(float lo, float hi) {
  unsigned int a = __float_as_uint(lo) + 0x8000u;
  unsigned int b = __float_as_uint(hi) + 0x8000u;
  return __builtin_amdgcn_perm(b, a, 0x07060302u);  // {b[31:16], a[31:16]}
}

// ---------------------------------------------------------------------------
// Kernel 0: one-time param conversion + transpose into workspace.
//   wbf_t : [64 n][2048 k] bf16   (w transposed, node 63 zero-padded)
//   rbf_t : [2048 c][64 k] bf16   (responses transposed)
// Hot loops then stage LDS tiles with pure b128 copies (no scatter).
// ---------------------------------------------------------------------------
__global__ __launch_bounds__(256) void hmoe_convert(
    const float* __restrict__ w, const float* __restrict__ resp,
    unsigned short* __restrict__ wbf_t, unsigned short* __restrict__ rbf_t)
{
  int i = blockIdx.x * 256 + threadIdx.x;      // 0 .. 131071
  {
    int n = i >> 11, k = i & 2047;
    float wv = (n < 63) ? w[k * 63 + n] : 0.0f;
    unsigned int bb = __float_as_uint(wv) + 0x8000u;
    wbf_t[i] = (unsigned short)(bb >> 16);
  }
  {
    int c = i >> 6, k = i & 63;
    unsigned int bb = __float_as_uint(resp[k * DIMY + c]) + 0x8000u;
    rbf_t[i] = (unsigned short)(bb >> 16);
  }
}

// ---------------------------------------------------------------------------
// Fused main kernel: 256 threads (8 wave32), 128 rows per block.
// ---------------------------------------------------------------------------
__global__ __launch_bounds__(256) void hmoe_fused(
    const float* __restrict__ x, const float* __restrict__ bias,
    const unsigned short* __restrict__ wbf_t, const unsigned short* __restrict__ rbf_t,
    float* __restrict__ out)
{
  // Strides: 40/72 shorts = 20/36 dwords -> lane b128 windows tile the 64 banks.
  __shared__ alignas(16) unsigned short xs[2][TM][40];   // x tile bf16 [row][k<32]
  __shared__ alignas(16) unsigned short wt[2][LVS][40];  // w tile [n][k<32]
  __shared__ alignas(16) float          gate_s[TM][65];  // sigmoid gates (pad 65)
  __shared__ alignas(16) unsigned short run_s[TM][72];   // running bf16 [row][leaf]
  __shared__ alignas(16) unsigned short rt[2][128][72];  // resp chunk [col][k<64]

  const int tid  = threadIdx.x;
  const int wave = tid >> 5;
  const int lane = tid & 31;
  const int half = lane >> 4;
  const int idx  = lane & 15;
  const int rb   = wave * 16;
  const long rowbase = (long)blockIdx.x * TM;

  // staging roles (phase 1)
  const int xrow = tid >> 1;              // x tile: 16 floats of one row
  const int xseg = (tid & 1) * 16;
  const int wn   = tid >> 2;              // w tile: 8 shorts of one node row
  const int wkq  = (tid & 3) * 8;

  // ---------------- Phase 1: gate logits = x @ w  (K = 2048, step 32) -------
  v8f acc[4];
#pragma unroll
  for (int t = 0; t < 4; ++t) acc[t] = {};

  unsigned int xr[8];
  uint4 wr;

  // prologue: fetch + stage tile 0
  {
    const float4* s = (const float4*)(x + (rowbase + xrow) * DIMX + xseg);
    float4 f0 = s[0], f1 = s[1], f2 = s[2], f3 = s[3];
    xr[0] = pack2(f0.x, f0.y); xr[1] = pack2(f0.z, f0.w);
    xr[2] = pack2(f1.x, f1.y); xr[3] = pack2(f1.z, f1.w);
    xr[4] = pack2(f2.x, f2.y); xr[5] = pack2(f2.z, f2.w);
    xr[6] = pack2(f3.x, f3.y); xr[7] = pack2(f3.z, f3.w);
    wr = *(const uint4*)(wbf_t + wn * DIMX + wkq);
    uint4* d = (uint4*)&xs[0][xrow][xseg];
    d[0] = make_uint4(xr[0], xr[1], xr[2], xr[3]);
    d[1] = make_uint4(xr[4], xr[5], xr[6], xr[7]);
    *(uint4*)&wt[0][wn][wkq] = wr;
  }
  __syncthreads();

  for (int i = 0; i < DIMX / 32; ++i) {
    const int cur = i & 1;
    const int kk2 = (i + 1) * 32;
    // issue global loads for tile i+1 (overlaps the WMMAs below)
    if (kk2 < DIMX) {
      const float4* s = (const float4*)(x + (rowbase + xrow) * DIMX + kk2 + xseg);
      float4 f0 = s[0], f1 = s[1], f2 = s[2], f3 = s[3];
      xr[0] = pack2(f0.x, f0.y); xr[1] = pack2(f0.z, f0.w);
      xr[2] = pack2(f1.x, f1.y); xr[3] = pack2(f1.z, f1.w);
      xr[4] = pack2(f2.x, f2.y); xr[5] = pack2(f2.z, f2.w);
      xr[6] = pack2(f3.x, f3.y); xr[7] = pack2(f3.z, f3.w);
      wr = *(const uint4*)(wbf_t + wn * DIMX + kk2 + wkq);
    }
    // compute from buf[cur]
    {
      Frag16 a;
      const unsigned short* ar = &xs[cur][rb + idx][0];
      a.q[0] = *(const uint4*)(ar + half * 8);
      a.q[1] = *(const uint4*)(ar + 16 + half * 8);
#pragma unroll
      for (int t = 0; t < 4; ++t) {
        Frag16 b;
        const unsigned short* br = &wt[cur][t * 16 + idx][0];
        b.q[0] = *(const uint4*)(br + half * 8);
        b.q[1] = *(const uint4*)(br + 16 + half * 8);
        acc[t] = __builtin_amdgcn_wmma_f32_16x16x32_bf16(
            false, a.v, false, b.v, (short)0, acc[t], false, false);
      }
    }
    // stage tile i+1 into buf[cur^1]
    if (kk2 < DIMX) {
      uint4* d = (uint4*)&xs[cur ^ 1][xrow][xseg];
      d[0] = make_uint4(xr[0], xr[1], xr[2], xr[3]);
      d[1] = make_uint4(xr[4], xr[5], xr[6], xr[7]);
      *(uint4*)&wt[cur ^ 1][wn][wkq] = wr;
    }
    __syncthreads();
  }

  // + bias, sigmoid, scatter to LDS.  C layout: (m = r + 8*half, n = idx).
#pragma unroll
  for (int t = 0; t < 4; ++t) {
    int n = t * 16 + idx;
    float bb = (n < 63) ? bias[n] : 0.0f;
#pragma unroll
    for (int r = 0; r < 8; ++r) {
      float v = acc[t][r] + bb;
      gate_s[rb + r + half * 8][n] = 1.0f / (1.0f + __expf(-v));
    }
  }
  __syncthreads();

  // ---------------- Phase 2: tree prefix products -> running (bf16) ---------
  // One thread per (row, half-of-leaves): 63 LDS loads, incremental products.
  {
    const int row = tid >> 1;
    const int b5  = tid & 1;     // top leaf bit (level 6 select)
    float g0 = gate_s[row][0];
    float r1[2]; r1[0] = g0; r1[1] = 1.0f - g0;
    float r2[4], r3[8], r4[16], r5[32];
#pragma unroll
    for (int m = 0; m < 4; ++m) {
      float g = gate_s[row][1 + (m & 1)];
      r2[m] = r1[m & 1] * ((m & 2) ? (1.0f - g) : g);
    }
#pragma unroll
    for (int m = 0; m < 8; ++m) {
      float g = gate_s[row][3 + (m & 3)];
      r3[m] = r2[m & 3] * ((m & 4) ? (1.0f - g) : g);
    }
#pragma unroll
    for (int m = 0; m < 16; ++m) {
      float g = gate_s[row][7 + (m & 7)];
      r4[m] = r3[m & 7] * ((m & 8) ? (1.0f - g) : g);
    }
#pragma unroll
    for (int m = 0; m < 32; ++m) {
      float g = gate_s[row][15 + (m & 15)];
      r5[m] = r4[m & 15] * ((m & 16) ? (1.0f - g) : g);
    }
    unsigned int pk[16];
#pragma unroll
    for (int j = 0; j < 32; j += 2) {
      float gA = gate_s[row][31 + j];
      float gB = gate_s[row][31 + j + 1];
      float vA = r5[j]     * (b5 ? (1.0f - gA) : gA);
      float vB = r5[j + 1] * (b5 ? (1.0f - gB) : gB);
      pk[j >> 1] = pack2(vA, vB);
    }
    uint4* d = (uint4*)&run_s[row][b5 * 32];
#pragma unroll
    for (int qq = 0; qq < 4; ++qq)
      d[qq] = make_uint4(pk[qq * 4], pk[qq * 4 + 1], pk[qq * 4 + 2], pk[qq * 4 + 3]);
  }
  __syncthreads();

  // ---------------- Phase 3: out = running @ responses  (K = 64) ------------
  Frag16 a2[2];
  {
    const unsigned short* ar = &run_s[rb + idx][0];
#pragma unroll
    for (int kc = 0; kc < 2; ++kc) {
      a2[kc].q[0] = *(const uint4*)(ar + kc * 32 + half * 8);
      a2[kc].q[1] = *(const uint4*)(ar + kc * 32 + 16 + half * 8);
    }
  }

  // staging roles (phase 3): each thread copies half a column (32 shorts)
  const int ccol = tid >> 1;
  const int ckh  = (tid & 1) * 32;
  uint4 rr[4];

  // prologue: fetch + stage chunk 0 (columns 0..127)
  {
    const uint4* s = (const uint4*)(rbf_t + (long)ccol * 64 + ckh);
#pragma unroll
    for (int q = 0; q < 4; ++q) rr[q] = s[q];
    uint4* d = (uint4*)&rt[0][ccol][ckh];
#pragma unroll
    for (int q = 0; q < 4; ++q) d[q] = rr[q];
  }
  __syncthreads();

  for (int c = 0; c < DIMY / 128; ++c) {
    const int cur = c & 1;
    const int nc2 = (c + 1) * 128;
    if (nc2 < DIMY) {
      const uint4* s = (const uint4*)(rbf_t + ((long)nc2 + ccol) * 64 + ckh);
#pragma unroll
      for (int q = 0; q < 4; ++q) rr[q] = s[q];
    }
#pragma unroll
    for (int nt = 0; nt < 8; ++nt) {
      Frag16 b0, b1;
      const unsigned short* br = &rt[cur][nt * 16 + idx][0];
      b0.q[0] = *(const uint4*)(br + half * 8);
      b0.q[1] = *(const uint4*)(br + 16 + half * 8);
      b1.q[0] = *(const uint4*)(br + 32 + half * 8);
      b1.q[1] = *(const uint4*)(br + 32 + 16 + half * 8);
      v8f o = {};
      o = __builtin_amdgcn_wmma_f32_16x16x32_bf16(
          false, a2[0].v, false, b0.v, (short)0, o, false, false);
      o = __builtin_amdgcn_wmma_f32_16x16x32_bf16(
          false, a2[1].v, false, b1.v, (short)0, o, false, false);
      long col = (long)c * 128 + nt * 16 + idx;
#pragma unroll
      for (int r = 0; r < 8; ++r)
        out[(rowbase + rb + r + half * 8) * DIMY + col] = o[r];
    }
    if (nc2 < DIMY) {
      uint4* d = (uint4*)&rt[cur ^ 1][ccol][ckh];
#pragma unroll
      for (int q = 0; q < 4; ++q) d[q] = rr[q];
    }
    __syncthreads();
  }
}

// ---------------------------------------------------------------------------
extern "C" void kernel_launch(void* const* d_in, const int* in_sizes, int n_in,
                              void* d_out, int out_size, void* d_ws, size_t ws_size,
                              hipStream_t stream) {
  (void)in_sizes; (void)n_in; (void)out_size; (void)ws_size;
  const float* x    = (const float*)d_in[0];   // [32768][2048]
  const float* w    = (const float*)d_in[1];   // [2048][63]
  const float* bias = (const float*)d_in[2];   // [63]
  const float* resp = (const float*)d_in[3];   // [64][2048]
  float* out = (float*)d_out;                  // [32768][2048]

  unsigned short* wbf_t = (unsigned short*)d_ws;       // [64][2048] bf16 (w^T)
  unsigned short* rbf_t = wbf_t + 64 * 2048;           // [2048][64] bf16 (resp^T)

  hipLaunchKernelGGL(hmoe_convert, dim3(512), dim3(256), 0, stream,
                     w, resp, wbf_t, rbf_t);
  hipLaunchKernelGGL(hmoe_fused, dim3(32768 / TM), dim3(256), 0, stream,
                     x, bias, wbf_t, rbf_t, out);
}
